// CrossAttentionModule_59184649338906
// MI455X (gfx1250) — matrix operand order repaired and checked by reading, not compiled, not run on hardware
//
#include <hip/hip_runtime.h>
#include <hip/hip_bf16.h>

typedef __attribute__((ext_vector_type(16))) __bf16 v16bf;
typedef __attribute__((ext_vector_type(2)))  __bf16 v2bf;
typedef __attribute__((ext_vector_type(8)))  float  v8f;
typedef __attribute__((ext_vector_type(4)))  unsigned int uint4v;

// ---------- bf16 conversion helpers (prefer HW packed cvt) ----------
__device__ __forceinline__ unsigned int pack_bf16x2(float lo, float hi) {
#if __has_builtin(__builtin_amdgcn_cvt_pk_bf16_f32)
  union { v2bf b; unsigned int u; } c;
  c.b = __builtin_amdgcn_cvt_pk_bf16_f32(lo, hi);
  return c.u;
#else
  union { v2bf b; unsigned int u; } c;
  c.b[0] = (__bf16)lo;
  c.b[1] = (__bf16)hi;
  return c.u;
#endif
}
__device__ __forceinline__ unsigned short f32_to_bf16_bits(float f) {
  union { __bf16 b; unsigned short u; } c;
  c.b = (__bf16)f;
  return c.u;
}

union frag32 { uint4v u[2]; v16bf b; };

// ---------- wide WMMA fragment loaders ----------
// A matrix 16x32 (MxK), row(M)-major k-fastest storage; STRIDE multiple of 8.
template <int STRIDE>
__device__ __forceinline__ v16bf load_frag_a_w(const unsigned short* s, int lane) {
  const unsigned short* p = s + (lane & 15) * STRIDE + ((lane >> 4) << 3);
  frag32 c;
  c.u[0] = *(const uint4v*)(p);
  c.u[1] = *(const uint4v*)(p + 16);
  return c.b;
}
// B matrix 32x16 (KxN), N-major k-fastest storage; STRIDE multiple of 8.
template <int STRIDE>
__device__ __forceinline__ v16bf load_frag_b_w(const unsigned short* s, int lane) {
  const unsigned short* p = s + (lane & 15) * STRIDE + ((lane >> 4) << 3);
  frag32 c;
  c.u[0] = *(const uint4v*)(p);
  c.u[1] = *(const uint4v*)(p + 16);
  return c.b;
}

// =====================================================================
// Kernel 1: QKV projection.  out[b,l,n] = sum_c x[b,c,l]*w[c,n] + bias[n]
// x: (B,C,L) f32   w: (C,512) f32   out: (B,L,512) bf16
// Block tile 128(M)x64(N), K-step 32; 8 waves as 4x2 of 32x32.
// =====================================================================
__global__ __launch_bounds__(256) void gemm_xt_w(
    const float* __restrict__ x, const float* __restrict__ w,
    const float* __restrict__ bias, unsigned short* __restrict__ out,
    int C, int L) {
  constexpr int TM = 128, TN = 64, TK = 32;
  constexpr int LDA = TK + 8;   // 40 ushorts -> 80B row stride (16B aligned)
  constexpr int LDB = TK + 8;   // Bs N-major: Bs[n][k]
  __shared__ __align__(16) unsigned short As[TM * LDA];
  __shared__ __align__(16) unsigned short Bs[TN * LDB];

  const int b  = blockIdx.z;
  const int m0 = blockIdx.x * TM;
  const int n0 = blockIdx.y * TN;
  const float* xb = x + (size_t)b * C * L;

  const int tid = threadIdx.x, lane = tid & 31, wave = tid >> 5;
  const int wm = (wave & 3) * 32, wn = (wave >> 2) * 32;

  // ---- per-thread staging geometry; pointers walk across the K loop ----
  const int jA0 = tid >> 7;              // 0..1 (k-pair block)
  const int mmA = tid & 127;
  const int gmA = m0 + mmA;
  const bool okA = gmA < L;
  const float* pa = xb + (size_t)(2 * jA0) * L + (okA ? gmA : (L - 1));
  unsigned int* dA = (unsigned int*)&As[mmA * LDA + 2 * jA0];

  const int jB0 = tid >> 6;              // 0..3
  const int nnB = tid & 63;
  const float* pb = w + (size_t)(2 * jB0) * 512 + n0 + nnB;
  unsigned int* dB = (unsigned int*)&Bs[nnB * LDB + 2 * jB0];

  v8f acc[2][2] = {};
  for (int k0 = 0; k0 < C; k0 += TK) {
#pragma unroll
    for (int i = 0; i < 8; ++i) {        // A: 2 k-pairs apart per step
      float v0 = pa[0];
      float v1 = pa[L];
      pa += (size_t)4 * L;
      dA[2 * i] = okA ? pack_bf16x2(v0, v1) : 0u;
    }
#pragma unroll
    for (int i = 0; i < 4; ++i) {        // B
      float v0 = pb[0];
      float v1 = pb[512];
      pb += 8 * 512;
      dB[4 * i] = pack_bf16x2(v0, v1);
    }
    __syncthreads();

    v16bf af[2], bf[2];
#pragma unroll
    for (int mi = 0; mi < 2; ++mi)
      af[mi] = load_frag_a_w<LDA>(&As[(wm + mi * 16) * LDA], lane);
#pragma unroll
    for (int ni = 0; ni < 2; ++ni)
      bf[ni] = load_frag_b_w<LDB>(&Bs[(wn + ni * 16) * LDB], lane);
#pragma unroll
    for (int mi = 0; mi < 2; ++mi)
#pragma unroll
      for (int ni = 0; ni < 2; ++ni)
        acc[mi][ni] = __builtin_amdgcn_wmma_f32_16x16x32_bf16(
            false, af[mi], false, bf[ni], (short)0, acc[mi][ni], false, false);
    __syncthreads();
  }

  // ---- epilogue: bias + bf16 store via one base pointer ----
  const int ncol = lane & 15, rowadd = (lane >> 4) << 3;
  unsigned short* po = out + ((size_t)b * L + (m0 + wm + rowadd)) * 512
                           + (n0 + wn + ncol);
  const float bv[2] = { bias[n0 + wn + ncol], bias[n0 + wn + 16 + ncol] };
#pragma unroll
  for (int mi = 0; mi < 2; ++mi)
#pragma unroll
    for (int r = 0; r < 8; ++r) {
      int gm = m0 + wm + mi * 16 + r + rowadd;
      if (gm < L) {
#pragma unroll
        for (int ni = 0; ni < 2; ++ni)
          po[(size_t)(mi * 16 + r) * 512 + ni * 16] =
              f32_to_bf16_bits(acc[mi][ni][r] + bv[ni]);
      }
    }
}

// =====================================================================
// Kernel 2: flash attention per (b, h).  q/k/v/o: (B,L,512) bf16.
// 8 waves; each wave owns 16 query rows (BR=128); BC=64 key blocks,
// online softmax, scale=1/8.  V staged transposed for b128 B-fragments.
// =====================================================================
__global__ __launch_bounds__(256) void flash_attn(
    const unsigned short* __restrict__ q, const unsigned short* __restrict__ k,
    const unsigned short* __restrict__ v, unsigned short* __restrict__ o,
    int L) {
  constexpr int BC = 64;
  constexpr int LDK = 64 + 8;   // Ks[key][d]
  constexpr int LDV = 64 + 8;   // Vt[d][key]
  constexpr int LDP = 64 + 8;   // Ps[row][key], per wave
  __shared__ __align__(16) unsigned short Ks[BC * LDK];
  __shared__ __align__(16) unsigned short Vt[64 * LDV];
  __shared__ __align__(16) unsigned short Ps[8 * 16 * LDP];

  const int b = blockIdx.z, h = blockIdx.y;
  const int m0 = blockIdx.x * 128;
  const int tid = threadIdx.x, lane = tid & 31, wave = tid >> 5;
  const size_t base = ((size_t)b * L) * 512 + (size_t)h * 64;

  // --- Q fragments straight from global (2x b128 each chunk) ---
  v16bf qf[2];
  {
    int row = m0 + wave * 16 + (lane & 15);
    bool ok = row < L;
    const unsigned short* qrow =
        q + base + (size_t)(ok ? row : 0) * 512 + ((lane >> 4) << 3);
    const uint4v zero = {};
#pragma unroll
    for (int c = 0; c < 2; ++c) {
      frag32 cc;
      cc.u[0] = ok ? *(const uint4v*)(qrow + c * 32)      : zero;
      cc.u[1] = ok ? *(const uint4v*)(qrow + c * 32 + 16) : zero;
      qf[c] = cc.b;
    }
  }

  // ---- staging geometry; K/V pointers walk across the key loop ----
  const int rrS = tid >> 3;              // 0..31
  const int d8S = (tid & 7) << 3;        // d octet
  const unsigned short* pk = k + base + (size_t)rrS * 512 + d8S;
  const unsigned short* pv = v + base + (size_t)rrS * 512 + d8S;
  uint4v* dK[2] = { (uint4v*)&Ks[rrS * LDK + d8S],
                    (uint4v*)&Ks[(rrS + 32) * LDK + d8S] };
  unsigned short* dV = &Vt[d8S * LDV + rrS];

  float m_run[8], l_run[8];
#pragma unroll
  for (int r = 0; r < 8; ++r) { m_run[r] = -1e30f; l_run[r] = 0.f; }
  v8f accO[4] = {};

  for (int kb = 0; kb < L; kb += BC) {
#pragma unroll
    for (int i = 0; i < 2; ++i) {
      *dK[i] = *(const uint4v*)(pk);
      union { uint4v u; unsigned short s[8]; } vv;
      vv.u = *(const uint4v*)(pv);
#pragma unroll
      for (int j = 0; j < 8; ++j) dV[(size_t)j * LDV + 32 * i] = vv.s[j];
      pk += 32 * 512;
      pv += 32 * 512;
    }
    __syncthreads();

    // ---- S = Q K^T ----
    v8f s[4] = {};
#pragma unroll
    for (int nt = 0; nt < 4; ++nt)
#pragma unroll
      for (int c = 0; c < 2; ++c) {
        v16bf bfr = load_frag_b_w<LDK>(&Ks[(nt * 16) * LDK + c * 32], lane);
        s[nt] = __builtin_amdgcn_wmma_f32_16x16x32_bf16(
            false, qf[c], false, bfr, (short)0, s[nt], false, false);
      }

    // ---- online softmax ----
    const float scale = 0.125f;
    unsigned short* ppw =
        &Ps[wave * 16 * LDP + ((lane >> 4) << 3) * LDP + (lane & 15)];
#pragma unroll
    for (int r = 0; r < 8; ++r) {
      float mx = m_run[r];
#pragma unroll
      for (int nt = 0; nt < 4; ++nt) { s[nt][r] *= scale; mx = fmaxf(mx, s[nt][r]); }
#pragma unroll
      for (int off = 8; off >= 1; off >>= 1)
        mx = fmaxf(mx, __shfl_xor(mx, off, 16));
      float sum = 0.f;
#pragma unroll
      for (int nt = 0; nt < 4; ++nt) {
        float p = __expf(s[nt][r] - mx);
        sum += p;
        ppw[r * LDP + nt * 16] = f32_to_bf16_bits(p);
      }
#pragma unroll
      for (int off = 8; off >= 1; off >>= 1)
        sum += __shfl_xor(sum, off, 16);
      float alpha = __expf(m_run[r] - mx);
      l_run[r] = l_run[r] * alpha + sum;
      m_run[r] = mx;
#pragma unroll
      for (int nt = 0; nt < 4; ++nt) accO[nt][r] *= alpha;
    }

    // ---- O += P @ V ----
    unsigned short* pw = &Ps[wave * 16 * LDP];
#pragma unroll
    for (int c = 0; c < 2; ++c) {
      v16bf pf = load_frag_a_w<LDP>(pw + c * 32, lane);
#pragma unroll
      for (int dt = 0; dt < 4; ++dt) {
        v16bf bfr = load_frag_b_w<LDV>(&Vt[(dt * 16) * LDV + c * 32], lane);
        accO[dt] = __builtin_amdgcn_wmma_f32_16x16x32_bf16(
            false, pf, false, bfr, (short)0, accO[dt], false, false);
      }
    }
    __syncthreads();
  }

  // ---- epilogue ----
  const int ncol = lane & 15, rowadd = (lane >> 4) << 3;
  unsigned short* po = o + base + (size_t)(m0 + wave * 16 + rowadd) * 512 + ncol;
#pragma unroll
  for (int r = 0; r < 8; ++r) {
    int gm = m0 + wave * 16 + r + rowadd;
    if (gm < L) {
      float inv = 1.f / l_run[r];
#pragma unroll
      for (int dt = 0; dt < 4; ++dt)
        po[(size_t)r * 512 + dt * 16] = f32_to_bf16_bits(accO[dt][r] * inv);
    }
  }
}

// =====================================================================
// Kernel 3: output projection + transpose-back.
// a: (B,L,512) bf16   w: (512,C) f32   out: (B,C,L) f32
// =====================================================================
__global__ __launch_bounds__(256) void gemm_o_w(
    const unsigned short* __restrict__ a, const float* __restrict__ w,
    const float* __restrict__ bias, float* __restrict__ out, int C, int L) {
  constexpr int TM = 128, TN = 64, TK = 32;
  constexpr int LDA = TK + 8, LDB = TK + 8;   // Bs N-major
  __shared__ __align__(16) unsigned short As[TM * LDA];
  __shared__ __align__(16) unsigned short Bs[TN * LDB];

  const int b  = blockIdx.z;
  const int m0 = blockIdx.x * TM;
  const int n0 = blockIdx.y * TN;
  const unsigned short* ab = a + (size_t)b * L * 512;

  const int tid = threadIdx.x, lane = tid & 31, wave = tid >> 5;
  const int wm = (wave & 3) * 32, wn = (wave >> 2) * 32;
  const uint4v zero = {};

  // ---- staging geometry ----
  const int mmA = tid >> 2;              // 0..63 (rows mmA and mmA+64)
  const int k8A = (tid & 3) << 3;
  const int gm0 = m0 + mmA, gm1 = m0 + mmA + 64;
  const bool ok0 = gm0 < L, ok1 = gm1 < L;
  const unsigned short* pa0 = ab + (size_t)(ok0 ? gm0 : (L - 1)) * 512 + k8A;
  const unsigned short* pa1 = ab + (size_t)(ok1 ? gm1 : (L - 1)) * 512 + k8A;
  uint4v* dA0 = (uint4v*)&As[mmA * LDA + k8A];
  uint4v* dA1 = (uint4v*)&As[(mmA + 64) * LDA + k8A];

  const int jB0 = tid >> 6;              // 0..3
  const int nnB = tid & 63;
  const float* pb = w + (size_t)(2 * jB0) * C + n0 + nnB;
  unsigned int* dB = (unsigned int*)&Bs[nnB * LDB + 2 * jB0];

  v8f acc[2][2] = {};
  for (int k0 = 0; k0 < 512; k0 += TK) {
    uint4v va0 = *(const uint4v*)(pa0 + k0);
    uint4v va1 = *(const uint4v*)(pa1 + k0);
    *dA0 = ok0 ? va0 : zero;
    *dA1 = ok1 ? va1 : zero;
#pragma unroll
    for (int i = 0; i < 4; ++i) {
      float v0 = pb[0];
      float v1 = pb[C];
      pb += (size_t)8 * C;
      dB[4 * i] = pack_bf16x2(v0, v1);
    }
    __syncthreads();

    v16bf af[2], bf[2];
#pragma unroll
    for (int mi = 0; mi < 2; ++mi)
      af[mi] = load_frag_a_w<LDA>(&As[(wm + mi * 16) * LDA], lane);
#pragma unroll
    for (int ni = 0; ni < 2; ++ni)
      bf[ni] = load_frag_b_w<LDB>(&Bs[(wn + ni * 16) * LDB], lane);
#pragma unroll
    for (int mi = 0; mi < 2; ++mi)
#pragma unroll
      for (int ni = 0; ni < 2; ++ni)
        acc[mi][ni] = __builtin_amdgcn_wmma_f32_16x16x32_bf16(
            false, af[mi], false, bf[ni], (short)0, acc[mi][ni], false, false);
    __syncthreads();
  }

  // ---- epilogue: transposed f32 store via base pointer ----
  const int ncol = lane & 15, rowadd = (lane >> 4) << 3;
  float* po = out + ((size_t)b * C + (n0 + wn + ncol)) * L + (m0 + wm + rowadd);
  const float bv[2] = { bias[n0 + wn + ncol], bias[n0 + wn + 16 + ncol] };
#pragma unroll
  for (int mi = 0; mi < 2; ++mi)
#pragma unroll
    for (int r = 0; r < 8; ++r) {
      int gm = m0 + wm + mi * 16 + r + rowadd;
      if (gm < L) {
#pragma unroll
        for (int ni = 0; ni < 2; ++ni)
          po[(size_t)ni * 16 * L + mi * 16 + r] = acc[mi][ni][r] + bv[ni];
      }
    }
}

// =====================================================================
// Host launcher
// =====================================================================
extern "C" void kernel_launch(void* const* d_in, const int* in_sizes, int n_in,
                              void* d_out, int out_size, void* d_ws, size_t ws_size,
                              hipStream_t stream) {
  (void)in_sizes; (void)n_in; (void)out_size; (void)ws_size;
  static const int DIMS[4] = {128, 256, 512, 512};
  static const int RES[4]  = {64, 32, 16, 8};
  const int Bb = 2;

  const size_t BUF = (size_t)Bb * 4096 * 512;     // elements per bf16 ws buffer
  unsigned short* qb = (unsigned short*)d_ws;
  unsigned short* kb = qb + BUF;
  unsigned short* vb = kb + BUF;
  unsigned short* ob = vb + BUF;

  float* outp = (float*)d_out;
  size_t out_off = 0;

  for (int i = 0; i < 4; ++i) {
    const int C = DIMS[i], R = RES[i], L = R * R;
    const int base = i * 11;
    const float* xq = (const float*)d_in[base + 0];   // ml_c
    const float* xk = (const float*)d_in[base + 1];   // ml_r
    const float* xv = (const float*)d_in[base + 2];   // fl_r
    const float* wq = (const float*)d_in[base + 3];
    const float* bq = (const float*)d_in[base + 4];
    const float* wk = (const float*)d_in[base + 5];
    const float* bk = (const float*)d_in[base + 6];
    const float* wv = (const float*)d_in[base + 7];
    const float* bv = (const float*)d_in[base + 8];
    const float* wo = (const float*)d_in[base + 9];
    const float* bo = (const float*)d_in[base + 10];

    dim3 blk(256);
    dim3 gproj((L + 127) / 128, 512 / 64, Bb);
    gemm_xt_w<<<gproj, blk, 0, stream>>>(xq, wq, bq, qb, C, L);
    gemm_xt_w<<<gproj, blk, 0, stream>>>(xk, wk, bk, kb, C, L);
    gemm_xt_w<<<gproj, blk, 0, stream>>>(xv, wv, bv, vb, C, L);

    dim3 gattn((L + 127) / 128, 8, Bb);
    flash_attn<<<gattn, blk, 0, stream>>>(qb, kb, vb, ob, L);

    dim3 gout((L + 127) / 128, C / 64, Bb);
    gemm_o_w<<<gout, blk, 0, stream>>>(ob, wo, bo, outp + out_off, C, L);

    out_off += (size_t)Bb * C * L;
  }
}
